// CUDAE80Cell_31138512896468
// MI455X (gfx1250) — compile-verified, matrix operand order checked by reading, not run on hardware
//
#include <hip/hip_runtime.h>
#include <hip/hip_bf16.h>

typedef __attribute__((ext_vector_type(16))) __bf16 v16bf;
typedef __attribute__((ext_vector_type(8)))  float  v8f;

#define T_DIM  2048
#define B_DIM  16
#define D_DIM  1024
#define N_DIM  64
#define PROJ_N 256   // 4*N stacked projections

// ---------------------------------------------------------------------------
// Fast sigmoid building block: sigmoid(z) = 0.5*(1 + tanh(z/2)); one trans op.
// ---------------------------------------------------------------------------
__device__ __forceinline__ float fast_tanh(float x) {
#if __has_builtin(__builtin_amdgcn_tanhf)
  return __builtin_amdgcn_tanhf(x);
#else
  float r;
  asm("v_tanh_f32 %0, %1" : "=v"(r) : "v"(x));
  return r;
#endif
}

__device__ __forceinline__ void sched_fence() {
#if __has_builtin(__builtin_amdgcn_sched_barrier)
  __builtin_amdgcn_sched_barrier(0);   // nothing crosses this point
#endif
}

// ---------------------------------------------------------------------------
// Kernel 0: pack W (256x1024 f32, row-major) into the WMMA B-fragment layout
// for V_WMMA_F32_16X16X32_BF16, split into bf16 hi + residual lo.
// Layout per (ntile, ktile): [lane 0..31][elem 0..15] where
//   lanes 0-15  : N = lane,    K = k0+{0..7, 16..23}
//   lanes 16-31 : N = lane-16, K = k0+{8..15, 24..31}
// (ISA 7.12.2, 16-bit matrix layout mirrored for B = KxN.)
// ---------------------------------------------------------------------------
__global__ void pack_w_kernel(const float* __restrict__ W,
                              __bf16* __restrict__ Whi,
                              __bf16* __restrict__ Wlo) {
  int idx = blockIdx.x * 256 + threadIdx.x;   // 0 .. 256*1024-1
  int n = idx >> 10;                          // W row (0..255)
  int k = idx & 1023;                         // W col (0..1023)
  float w  = W[idx];
  __bf16 hi = (__bf16)w;
  __bf16 lo = (__bf16)(w - (float)hi);
  int ntile = n >> 4, nl = n & 15;
  int ktile = k >> 5, kk = k & 31;
  int lane = nl + ((kk & 8)  ? 16 : 0);
  int e    = (kk & 7) + ((kk & 16) ? 8 : 0);
  long dst = (((long)(ntile * 32 + ktile) * 32 + lane) * 16 + e);
  Whi[dst] = hi;
  Wlo[dst] = lo;
}

// ---------------------------------------------------------------------------
// Kernel 1: kvqm[m][n] = sum_d X[m][d] * W[n][d]  (m = t*16+b, 32768x256)
// 512 threads = 16 waves per block; block owns a 16-row M strip, each wave
// owns one 16-wide N tile. 64-wide K tiles staged in LDS. All operand
// fragments are built first, then 6 WMMAs issue as one sched-fenced block:
// accumulator reuse distance is 3 and no VALU overwrites WMMA sources
// inside the §7.12.1 hazard window -> no v_nop bubbles.
// ---------------------------------------------------------------------------
__global__ __launch_bounds__(512) void proj_gemm_kernel(
    const float* __restrict__ X,
    const __bf16* __restrict__ Whi,
    const __bf16* __restrict__ Wlo,
    float* __restrict__ kvqm) {
  __shared__ float ldsA[16 * 64];   // 4 KB K-tile of the X strip

  const int  tid  = threadIdx.x;
  const int  wave = tid >> 5;          // 0..15 -> N tile index
  const int  lane = tid & 31;
  const long m0   = (long)blockIdx.x * 16;
  const int  arow = lane & 15;                 // A-matrix M row for this lane
  const int  kb   = (lane >> 4) ? 8 : 0;       // lane-group K base offset

  // staging addresses: each thread moves one float2 (coalesced b64)
  const int srow = tid >> 5;          // 0..15
  const int scol = (tid & 31) * 2;    // 0,2,..,62

  v8f acc_hh = {}, acc_hl = {}, acc_lh = {};

  for (int k0 = 0; k0 < D_DIM; k0 += 64) {
    __syncthreads();   // previous tile's readers done
    {
      const float* src = &X[(m0 + srow) * D_DIM + k0 + scol];
      *(float2*)&ldsA[srow * 64 + scol] = *(const float2*)src;
      if (k0 + 64 < D_DIM) {
        // warm L2/WGP$ for the next strip tile (global_prefetch_b8)
        __builtin_prefetch(src + 64, 0, 0);
      }
    }
    __syncthreads();

    // ---- build ALL fragments for this 64-wide K tile up front ----
    v16bf ahi[2], alo[2], bhi[2], blo[2];
#pragma unroll
    for (int kt = 0; kt < 2; ++kt) {
      const float* lrow = &ldsA[arow * 64 + kt * 32];
      float a[16];
#pragma unroll
      for (int e = 0; e < 8; ++e) a[e]     = lrow[kb + e];
#pragma unroll
      for (int e = 0; e < 8; ++e) a[8 + e] = lrow[kb + 16 + e];
#pragma unroll
      for (int e = 0; e < 16; ++e) {
        __bf16 h = (__bf16)a[e];
        ahi[kt][e] = h;
        alo[kt][e] = (__bf16)(a[e] - (float)h);
      }
      long boff = (((long)(wave * 32 + ((k0 >> 5) + kt)) * 32 + lane) * 16);
      bhi[kt] = *(const v16bf*)(Whi + boff);
      blo[kt] = *(const v16bf*)(Wlo + boff);
    }

    // ---- 6 back-to-back WMMAs, independent accumulators, sched-fenced ----
    sched_fence();
    acc_hh = __builtin_amdgcn_wmma_f32_16x16x32_bf16(
        false, ahi[0], false, bhi[0], (short)0, acc_hh, false, false);
    acc_hl = __builtin_amdgcn_wmma_f32_16x16x32_bf16(
        false, ahi[0], false, blo[0], (short)0, acc_hl, false, false);
    acc_lh = __builtin_amdgcn_wmma_f32_16x16x32_bf16(
        false, alo[0], false, bhi[0], (short)0, acc_lh, false, false);
    acc_hh = __builtin_amdgcn_wmma_f32_16x16x32_bf16(
        false, ahi[1], false, bhi[1], (short)0, acc_hh, false, false);
    acc_hl = __builtin_amdgcn_wmma_f32_16x16x32_bf16(
        false, ahi[1], false, blo[1], (short)0, acc_hl, false, false);
    acc_lh = __builtin_amdgcn_wmma_f32_16x16x32_bf16(
        false, alo[1], false, bhi[1], (short)0, acc_lh, false, false);
    sched_fence();
  }

  v8f acc;
#pragma unroll
  for (int r = 0; r < 8; ++r) acc[r] = (acc_hh[r] + acc_hl[r]) + acc_lh[r];

  // C layout: VGPR r, lanes 0-15: M=r, N=lane; lanes 16-31: M=8+r, N=lane-16
  const int crow = (lane >> 4) ? 8 : 0;
  const int col  = lane & 15;
  const int n0   = wave * 16;
#pragma unroll
  for (int r = 0; r < 8; ++r) {
    kvqm[(m0 + crow + r) * PROJ_N + n0 + col] = acc[r];
  }
}

// ---------------------------------------------------------------------------
// Kernel 2: sequential gated scan. One block per batch; 256 threads =
// 64 rows x 4 column-groups of 16. S/M state lives in VGPRs for all 2048
// steps. kvqm rows double-buffered in LDS (1 barrier/step). Row matvec
// reductions: 2 shfl_xor hops (4 adjacent lanes per row in wave32).
// ---------------------------------------------------------------------------
__global__ __launch_bounds__(256) void scan_kernel(
    const float* __restrict__ kvqm,
    const float* __restrict__ BS,
    const float* __restrict__ BM,
    const float* __restrict__ S0,
    const float* __restrict__ M0,
    float* __restrict__ outs,   // (T, B, N)
    float* __restrict__ Sf,     // (B, N, N)
    float* __restrict__ Mf) {   // (B, N, N)
  __shared__ float buf[2][PROJ_N];

  const int b   = blockIdx.x;
  const int tid = threadIdx.x;
  const int row = tid >> 2;        // 0..63
  const int g   = tid & 3;         // column group
  const int j0  = g * 16;

  float S[16], M[16], Bs[16], Bm[16];
  const long st0 = (long)b * N_DIM * N_DIM + row * N_DIM + j0;
#pragma unroll
  for (int j = 0; j < 16; ++j) {
    Bs[j] = BS[row * N_DIM + j0 + j];
    Bm[j] = BM[row * N_DIM + j0 + j];
    S[j]  = S0[st0 + j];
    M[j]  = M0[st0 + j];
  }

  // preload step 0 (one float per thread)
  buf[0][tid] = kvqm[(long)b * PROJ_N + tid];

  for (int t = 0; t < T_DIM; ++t) {
    __syncthreads();              // buf[t&1] stores visible; buf[(t+1)&1] free
    const int cur = t & 1;
    if (t + 1 < T_DIM) {          // prefetch next step into the other buffer
      const float* nxt = &kvqm[((long)(t + 1) * B_DIM + b) * PROJ_N + tid];
      buf[cur ^ 1][tid] = *nxt;
      if (t + 2 < T_DIM) {
        // keep t+2's row moving toward the WGP (global_prefetch_b8)
        __builtin_prefetch(nxt + (long)B_DIM * PROJ_N, 0, 0);
      }
    }

    const float vrow = buf[cur][64 + row];
    float kj[16], qj[16], mj[16];
#pragma unroll
    for (int j = 0; j < 16; ++j) {
      kj[j] = buf[cur][j0 + j];            // k: cols 0..63
      qj[j] = buf[cur][128 + j0 + j];      // q: cols 128..191
      mj[j] = buf[cur][192 + j0 + j];      // m: cols 192..255
    }

    // S update + first-level read  (sigmoid(z) = 0.5 + 0.5*tanh(z/2))
    float part = 0.f;
#pragma unroll
    for (int j = 0; j < 16; ++j) {
      float vk = vrow * kj[j];
      float th = fast_tanh(0.5f * (vk + Bs[j]));
      S[j] = 0.5f * (S[j] + vk) + 0.5f * th * (S[j] - vk);
      part += S[j] * qj[j];
    }
    part += __shfl_xor(part, 1, 32);
    part += __shfl_xor(part, 2, 32);       // all 4 lanes now hold Sq[row]

    // M update + second-level read
    float outp = 0.f;
#pragma unroll
    for (int j = 0; j < 16; ++j) {
      float sm = part * mj[j];
      float th = fast_tanh(0.5f * (sm + Bm[j]));
      M[j] = 0.5f * (M[j] + sm) + 0.5f * th * (M[j] - sm);
      outp += M[j] * qj[j];
    }
    outp += __shfl_xor(outp, 1, 32);
    outp += __shfl_xor(outp, 2, 32);

    if (g == 0) outs[((long)t * B_DIM + b) * N_DIM + row] = outp;
  }

  // final states
#pragma unroll
  for (int j = 0; j < 16; ++j) {
    Sf[st0 + j] = S[j];
    Mf[st0 + j] = M[j];
  }
}

// ---------------------------------------------------------------------------
extern "C" void kernel_launch(void* const* d_in, const int* in_sizes, int n_in,
                              void* d_out, int out_size, void* d_ws, size_t ws_size,
                              hipStream_t stream) {
  const float* x  = (const float*)d_in[0];   // (T, B, D)
  const float* S0 = (const float*)d_in[1];   // (B, N, N)
  const float* M0 = (const float*)d_in[2];   // (B, N, N)
  const float* W  = (const float*)d_in[3];   // (4N, D)
  const float* BS = (const float*)d_in[4];   // (N, N)
  const float* BM = (const float*)d_in[5];   // (N, N)

  float* out = (float*)d_out;

  // workspace: kvqm f32 (32 MB) | Whi bf16 (512 KB) | Wlo bf16 (512 KB)
  float*  kvqm = (float*)d_ws;
  __bf16* Whi  = (__bf16*)((char*)d_ws + (size_t)T_DIM * B_DIM * PROJ_N * 4);
  __bf16* Wlo  = Whi + (size_t)PROJ_N * D_DIM;

  pack_w_kernel<<<(PROJ_N * D_DIM) / 256, 256, 0, stream>>>(W, Whi, Wlo);
  proj_gemm_kernel<<<(T_DIM * B_DIM) / 16, 512, 0, stream>>>(x, Whi, Wlo, kvqm);

  float* outs = out;
  float* Sf   = out + (long)T_DIM * B_DIM * N_DIM;
  float* Mf   = Sf + (long)B_DIM * N_DIM * N_DIM;
  scan_kernel<<<B_DIM, 256, 0, stream>>>(kvqm, BS, BM, S0, M0, outs, Sf, Mf);
}
